// Model_77154792505362
// MI455X (gfx1250) — matrix operand order repaired and checked by reading, not compiled, not run on hardware
//
#include <hip/hip_runtime.h>
#include <cstdint>
#include <cstddef>

// ---------------------------------------------------------------------------
// Problem constants (from reference): N=6144 nodes, F_IN=256, D=512, F_OUT=256
// ---------------------------------------------------------------------------
#define NDIM  6144
#define FIN   256
#define DDIM  512
#define FOUT  256
#define NEG_SLOPE 0.1f

typedef __attribute__((ext_vector_type(16))) _Float16 v16h;
typedef __attribute__((ext_vector_type(8)))  _Float16 v8h;
typedef __attribute__((ext_vector_type(8)))  float    v8f;

// ---------------------------------------------------------------------------
// WMMA helpers (CDNA5 wave32, V_WMMA_F32_16X16X32_F16)
//
// A operand (16x32 f16): lane m = lane%16 holds row m.
//   lanes 0-15 : halves 0-7 -> K = kk+0..7,  halves 8-15 -> K = kk+16..23
//   lanes 16-31: halves 0-7 -> K = kk+8..15, halves 8-15 -> K = kk+24..31
// B operand (32x16 f16): lane n = lane%16 holds column n.
//   lanes 0-15 -> K = kk+0..7,16..23, lanes 16-31 -> K = kk+8..15,24..31 (same order as A).
// C/D (16x16 f32): lane holds column n = lane%16, rows m = r + 8*(lane/16).
// ---------------------------------------------------------------------------
__device__ __forceinline__ v8f wmma32(v16h a, v16h b, v8f c) {
  return __builtin_amdgcn_wmma_f32_16x16x32_f16(
      /*neg_a=*/false, a, /*neg_b=*/false, b,
      /*c_mod=*/(short)0, c, /*reuse_a=*/false, /*reuse_b=*/false);
}

// base -> element (row 0, k kk) of a row-major 16 x (>=32) tile, leading dim ld
__device__ __forceinline__ v16h load_a_frag(const _Float16* base, int ld) {
  const int lane = threadIdx.x & 31;
  const int m    = lane & 15;
  const int hi   = lane >> 4;
  const _Float16* p = base + (size_t)m * ld + hi * 8;
  v8h lo = *(const v8h*)(p);        // K = hi*8 + 0..7
  v8h up = *(const v8h*)(p + 16);   // K = 16 + hi*8 + 0..7
  return __builtin_shufflevector(lo, up, 0,1,2,3,4,5,6,7,8,9,10,11,12,13,14,15);
}

// base -> element (col 0, k kk) where columns are stored as rows (len-K rows),
// leading dim ld.  Lane n reads two 8-half runs of its column (K = hi*8.., 16+hi*8..).
__device__ __forceinline__ v16h load_b_frag(const _Float16* base, int ld) {
  const int lane = threadIdx.x & 31;
  const int n    = lane & 15;
  const int hi   = lane >> 4;
  const _Float16* p = base + (size_t)n * ld + hi * 8;
  v8h lo = *(const v8h*)(p);
  v8h up = *(const v8h*)(p + 16);
  return __builtin_shufflevector(lo, up, 0,1,2,3,4,5,6,7,8,9,10,11,12,13,14,15);
}

// Fast tanh: prefer the gfx1250 V_TANH_F32 transcendental when exposed.
__device__ __forceinline__ float fast_tanhf(float x) {
#if __has_builtin(__builtin_amdgcn_tanhf)
  return __builtin_amdgcn_tanhf(x);
#else
  return tanhf(x);
#endif
}

// CDNA5 async copy global -> LDS (ASYNCcnt-tracked), 16 bytes per lane.
__device__ __forceinline__ void async_load_b128(uint32_t lds_off,
                                                const void* gptr) {
  asm volatile("global_load_async_to_lds_b128 %0, %1, off"
               :
               : "v"(lds_off), "v"((uint64_t)(uintptr_t)gptr)
               : "memory");
}
__device__ __forceinline__ void wait_asynccnt_le2() {
  asm volatile("s_wait_asynccnt 0x2" ::: "memory");
}
__device__ __forceinline__ void wait_asynccnt_0() {
  asm volatile("s_wait_asynccnt 0x0" ::: "memory");
}

// ---------------------------------------------------------------------------
// Elementwise conversions
// ---------------------------------------------------------------------------
__global__ void cvt_f16_kernel(const float* __restrict__ in,
                               _Float16* __restrict__ out, int n) {
  int i = blockIdx.x * blockDim.x + threadIdx.x;
  if (i < n) out[i] = (_Float16)in[i];
}

// in: [K x Nc] f32 row-major -> out: [Nc x K] f16 (weights: small, one-shot)
__global__ void cvt_transpose_f16_kernel(const float* __restrict__ in,
                                         _Float16* __restrict__ out,
                                         int K, int Nc) {
  int idx = blockIdx.x * blockDim.x + threadIdx.x;
  if (idx < K * Nc) {
    int k = idx / Nc, n = idx % Nc;
    out[(size_t)n * K + k] = (_Float16)in[idx];
  }
}

// LDS-tiled f16 transpose: in [M x Nc] -> out [Nc x M]
__global__ void __launch_bounds__(256)
transpose_h_kernel(const _Float16* __restrict__ in, _Float16* __restrict__ out,
                   int M, int Nc) {
  __shared__ _Float16 tile[32][33];
  int bx = blockIdx.x * 32;   // column tile (Nc)
  int by = blockIdx.y * 32;   // row tile (M)
  for (int i = threadIdx.y; i < 32; i += 8)
    tile[i][threadIdx.x] = in[(size_t)(by + i) * Nc + bx + threadIdx.x];
  __syncthreads();
  for (int i = threadIdx.y; i < 32; i += 8)
    out[(size_t)(bx + i) * M + by + threadIdx.x] = tile[threadIdx.x][i];
}

// ---------------------------------------------------------------------------
// Row 2-norm of h16 [NDIM x DDIM]; one wave32 per row
// ---------------------------------------------------------------------------
__global__ void __launch_bounds__(256)
rownorm_kernel(const _Float16* __restrict__ h, float* __restrict__ norms) {
  int row  = blockIdx.x * 8 + (threadIdx.x >> 5);
  int lane = threadIdx.x & 31;
  const _Float16* p = h + (size_t)row * DDIM + lane * 16;
  v8h a = *(const v8h*)p;
  v8h b = *(const v8h*)(p + 8);
  float s = 0.f;
#pragma unroll
  for (int i = 0; i < 8; ++i) {
    float x0 = (float)a[i], x1 = (float)b[i];
    s += x0 * x0 + x1 * x1;
  }
#pragma unroll
  for (int off = 16; off > 0; off >>= 1) s += __shfl_xor(s, off, 32);
  if (lane == 0) norms[row] = sqrtf(s);
}

// ---------------------------------------------------------------------------
// WMMA GEMM: out = act(A16[M,K] @ Bt16[N,K]^T + bias)
// Block = 128 threads = 4 waves; each wave: 16 rows x 64 cols (4 accumulators).
// ---------------------------------------------------------------------------
template <bool LEAKY, bool STORE_F32>
__global__ void __launch_bounds__(128)
gemm16_kernel(const _Float16* __restrict__ A, const _Float16* __restrict__ Bt,
              const float* __restrict__ bias, _Float16* __restrict__ out16,
              float* __restrict__ out32, int M, int Nout, int K) {
  const int lane = threadIdx.x & 31;
  const int w    = threadIdx.x >> 5;
  const int m0   = blockIdx.x * 16;
  const int n0   = blockIdx.y * 256 + w * 64;
  if (n0 >= Nout) return;

  v8f acc[4];
#pragma unroll
  for (int f = 0; f < 4; ++f) acc[f] = v8f{};

  for (int kk = 0; kk < K; kk += 32) {
    v16h a = load_a_frag(A + (size_t)m0 * K + kk, K);
#pragma unroll
    for (int f = 0; f < 4; ++f) {
      v16h b = load_b_frag(Bt + (size_t)(n0 + f * 16) * K + kk, K);
      acc[f] = wmma32(a, b, acc[f]);
    }
  }

  const int n_l = lane & 15;
  const int mb  = (lane >> 4) * 8;
#pragma unroll
  for (int f = 0; f < 4; ++f) {
    int n    = n0 + f * 16 + n_l;
    float bv = bias[n];
#pragma unroll
    for (int r = 0; r < 8; ++r) {
      float v = acc[f][r] + bv;
      if (LEAKY) v = (v >= 0.f) ? v : NEG_SLOPE * v;
      size_t idx = (size_t)(m0 + mb + r) * Nout + n;
      if (STORE_F32) out32[idx] = v;
      else           out16[idx] = (_Float16)v;
    }
  }
}

// ---------------------------------------------------------------------------
// Fused affinity + row-normalize + aggregate (flash-style single pass):
//   w_ij = exp(tanh(cos(h_i,h_j)*gb/8)*8) * A[i,j]
//   hagg_i = (sum_j w_ij h_j) / (sum_j w_ij)
// Block = 128 threads (4 waves) per 16-row i-tile.  j-tile = 64:
//   wave w computes S cols [w*16, w*16+16) and aggregates D cols [w*128, +128).
// The 151 MB adjacency stream is double-buffered into LDS with CDNA5
// global_load_async_to_lds_b128 (ASYNCcnt), overlapping HBM with WMMA.
// No S/Aff matrix is ever materialized (saves ~300 MB of HBM traffic).
// ---------------------------------------------------------------------------
__global__ void __launch_bounds__(128)
affinity_agg_kernel(const _Float16* __restrict__ h16,
                    const _Float16* __restrict__ hT16,
                    const float* __restrict__ norms,
                    const float* __restrict__ Aadj,
                    const float* __restrict__ gb,
                    _Float16* __restrict__ hagg16) {
  __shared__ _Float16 sHi[16 * DDIM];             // h_i tile (16 KB)
  __shared__ _Float16 sW[16 * 64];                // affinity weight tile (f16)
  __shared__ __align__(16) float sA[2][16 * 64];  // adjacency double buffer
  __shared__ float    sRowSum[4][16];
  __shared__ float    sTot[16];
  __shared__ float    sNormI[16];

  const int tid  = threadIdx.x;
  const int lane = tid & 31;
  const int w    = tid >> 5;
  const int i0   = blockIdx.x * 16;
  const float gbv = gb[0];

  // Stage h_i rows into LDS (16*512 halves; 64 halves per thread).
  {
    const _Float16* src = h16 + (size_t)i0 * DDIM;
#pragma unroll
    for (int it = 0; it < 8; ++it) {
      int idx = (it * 128 + tid) * 8;
      *(v8h*)(sHi + idx) = *(const v8h*)(src + idx);
    }
    if (tid < 16) sNormI[tid] = norms[i0 + tid];
  }

  // Async-stage this wave's 16x16 f32 chunk of A[i0..i0+15][jc..jc+15]
  // into sA[buf]; 2 x b128-per-lane instructions per tile (wave-local).
  auto stage_A_tile = [&](int jc, int buf) {
#pragma unroll
    for (int it = 0; it < 2; ++it) {
      int e = it * 128 + lane * 4;       // element within 16x16 chunk
      int m = e >> 4;
      int c = e & 15;
      const float* gp = Aadj + (size_t)(i0 + m) * NDIM + jc + c;
      uint32_t loff =
          (uint32_t)(uintptr_t)(&sA[buf][m * 64 + w * 16 + c]);
      async_load_b128(loff, gp);
    }
  };

  // Prologue: stage tile 0.
  stage_A_tile(w * 16, 0);

  __syncthreads();

  const int n_l = lane & 15;
  const int mb  = (lane >> 4) * 8;

  v8f accN[8];                 // 16 x 128 f32 numerator slice per wave
#pragma unroll
  for (int f = 0; f < 8; ++f) accN[f] = v8f{};
  float rs[8];
#pragma unroll
  for (int r = 0; r < 8; ++r) rs[r] = 0.f;

  for (int j0 = 0; j0 < NDIM; j0 += 64) {
    const int jc  = j0 + w * 16;
    const int buf = (j0 >> 6) & 1;

    // Kick off the next adjacency tile before doing this tile's compute.
    if (j0 + 64 < NDIM) stage_A_tile(jc + 64, buf ^ 1);

    // ---- S tile: (h_i * h_j^T) over K = DDIM ----
    v8f sAcc = v8f{};
    for (int kk = 0; kk < DDIM; kk += 32) {
      v16h a = load_a_frag(sHi + kk, DDIM);                       // LDS
      v16h b = load_b_frag(h16 + (size_t)jc * DDIM + kk, DDIM);   // L2-resident
      sAcc = wmma32(a, b, sAcc);
    }

    // Current A tile landed (in-order async completion; next may be in flight).
    if (j0 + 64 < NDIM) wait_asynccnt_le2();
    else                wait_asynccnt_0();

    __syncthreads();  // previous iteration's aggregation readers of sW done

    // ---- epilogue: cosine, tanh, exp, mask by A, accumulate row sums ----
    {
      float njv = norms[jc + n_l];
#pragma unroll
      for (int r = 0; r < 8; ++r) {
        int   m   = mb + r;
        float niv = sNormI[m];
        float s   = sAcc[r] / (niv * njv + 1e-20f);
        float t   = fast_tanhf(s * gbv * 0.125f);
        float wv  = __expf(t * 8.0f) * sA[buf][m * 64 + w * 16 + n_l];
        rs[r] += wv;
        sW[m * 64 + w * 16 + n_l] = (_Float16)wv;
      }
    }
    __syncthreads();  // sW complete (all 64 j-columns)

    // ---- aggregate: accN += W(16x64) @ h_j(64 x 128-slice) ----
#pragma unroll
    for (int kk2 = 0; kk2 < 64; kk2 += 32) {
      v16h a = load_a_frag(sW + kk2, 64);                         // LDS
#pragma unroll
      for (int f = 0; f < 8; ++f) {
        int d  = w * 128 + f * 16;
        v16h b = load_b_frag(hT16 + (size_t)d * NDIM + j0 + kk2, NDIM);
        accN[f] = wmma32(a, b, accN[f]);
      }
    }
  }

  // ---- row-sum reduction across the 16 lanes of each half-wave ----
#pragma unroll
  for (int r = 0; r < 8; ++r) {
    float v = rs[r];
#pragma unroll
    for (int off = 1; off < 16; off <<= 1) v += __shfl_xor(v, off, 32);
    rs[r] = v;
  }
  if (n_l == 0) {
#pragma unroll
    for (int r = 0; r < 8; ++r) sRowSum[w][mb + r] = rs[r];
  }
  __syncthreads();
  if (tid < 16)
    sTot[tid] = sRowSum[0][tid] + sRowSum[1][tid] +
                sRowSum[2][tid] + sRowSum[3][tid];
  __syncthreads();

  // ---- normalize and write the aggregated rows (f16 for next GEMM) ----
#pragma unroll
  for (int f = 0; f < 8; ++f) {
    int d = w * 128 + f * 16 + n_l;
#pragma unroll
    for (int r = 0; r < 8; ++r) {
      int   m = mb + r;
      float v = accN[f][r] / (sTot[m] + 1e-20f);
      hagg16[(size_t)(i0 + m) * DDIM + d] = (_Float16)v;
    }
  }
}

// ---------------------------------------------------------------------------
// Host launcher
// ---------------------------------------------------------------------------
extern "C" void kernel_launch(void* const* d_in, const int* in_sizes, int n_in,
                              void* d_out, int out_size, void* d_ws,
                              size_t ws_size, hipStream_t stream) {
  (void)in_sizes; (void)n_in; (void)out_size; (void)ws_size;
  const float* x  = (const float*)d_in[0];
  const float* A  = (const float*)d_in[1];
  const float* W1 = (const float*)d_in[2];
  const float* b1 = (const float*)d_in[3];
  const float* gb = (const float*)d_in[4];
  const float* Wg = (const float*)d_in[5];
  const float* bg = (const float*)d_in[6];
  const float* W2 = (const float*)d_in[7];
  const float* b2 = (const float*)d_in[8];
  float* out = (float*)d_out;

  uint8_t* ws = (uint8_t*)d_ws;
  size_t off = 0;
  auto alloc = [&](size_t bytes) -> void* {
    void* p = ws + off;
    off += (bytes + 255) & ~(size_t)255;
    return p;
  };
  _Float16* x16    = (_Float16*)alloc((size_t)NDIM * FIN  * 2);
  _Float16* W1t    = (_Float16*)alloc((size_t)DDIM * FIN  * 2);
  _Float16* Wgt    = (_Float16*)alloc((size_t)DDIM * DDIM * 2);
  _Float16* W2t    = (_Float16*)alloc((size_t)FOUT * DDIM * 2);
  _Float16* h16    = (_Float16*)alloc((size_t)NDIM * DDIM * 2);
  _Float16* hT16   = (_Float16*)alloc((size_t)DDIM * NDIM * 2);
  _Float16* hagg16 = (_Float16*)alloc((size_t)NDIM * DDIM * 2);
  _Float16* h2_16  = (_Float16*)alloc((size_t)NDIM * DDIM * 2);
  float*    norms  = (float*)  alloc((size_t)NDIM * 4);

  // precision prep: x -> f16 rows; weights -> f16 transposed ([N_out, K])
  { int n = NDIM * FIN;
    cvt_f16_kernel<<<(n + 255) / 256, 256, 0, stream>>>(x, x16, n); }
  { int n = FIN * DDIM;
    cvt_transpose_f16_kernel<<<(n + 255) / 256, 256, 0, stream>>>(W1, W1t, FIN, DDIM); }
  { int n = DDIM * DDIM;
    cvt_transpose_f16_kernel<<<(n + 255) / 256, 256, 0, stream>>>(Wg, Wgt, DDIM, DDIM); }
  { int n = DDIM * FOUT;
    cvt_transpose_f16_kernel<<<(n + 255) / 256, 256, 0, stream>>>(W2, W2t, DDIM, FOUT); }

  // h = leaky(x @ W1 + b1)
  gemm16_kernel<true, false><<<dim3(NDIM / 16, (DDIM + 255) / 256), 128, 0, stream>>>(
      x16, W1t, b1, h16, nullptr, NDIM, DDIM, FIN);

  // row norms + transposed copy of h for the aggregation B-operand
  rownorm_kernel<<<NDIM / 8, 256, 0, stream>>>(h16, norms);
  transpose_h_kernel<<<dim3(DDIM / 32, NDIM / 32), dim3(32, 8), 0, stream>>>(
      h16, hT16, NDIM, DDIM);

  // fused cosine-affinity softmax-like aggregation (no S/Aff materialization)
  affinity_agg_kernel<<<NDIM / 16, 128, 0, stream>>>(h16, hT16, norms, A, gb, hagg16);

  // h2 = leaky(hagg @ Wg + bg)
  gemm16_kernel<true, false><<<dim3(NDIM / 16, (DDIM + 255) / 256), 128, 0, stream>>>(
      hagg16, Wgt, bg, h2_16, nullptr, NDIM, DDIM, DDIM);

  // out = h2 @ W2 + b2  (f32 output)
  gemm16_kernel<false, true><<<dim3(NDIM / 16, (FOUT + 255) / 256), 128, 0, stream>>>(
      h2_16, W2t, b2, nullptr, out, NDIM, FOUT, DDIM);
}
